// Hgru2dModule_62603443306630
// MI455X (gfx1250) — matrix operand order repaired and checked by reading, not compiled, not run on hardware
//
#include <hip/hip_runtime.h>
#include <hip/hip_bf16.h>

// ---------------------------------------------------------------------------
// HGRU-2D for MI455X (gfx1250, wave32, WMMA + Tensor Data Mover).
//
//   0) one-time fp32 -> (hi,lo) bf16 split of x and of all four weight
//      matrices (weights also transposed to [n][k] so B fragments are
//      contiguous-K); removes ALL conversion VALU from the GEMM hot loop.
//   1) gemm_bf16pre x3 : U=silu(x@W_in+b), F=sigmoid(x@W_f+b), G=sigmoid(x@W_g+b)
//      inner loop = TDM tile DMA (double buffered, s_wait_tensorcnt) +
//      ds_load_b128 fragments + 12x v_wmma_f32_16x16x32_bf16 per k-chunk
//      (hi*hi + hi*lo + lo*hi split recovers ~fp32 accuracy).
//   2) scan2d (axis=w, init) + scan2d (axis=h, accumulate): 4 directional
//      complex gated scans, lam/(1-lam)u computed on the fly.
//   3) rownorm + split of (G * hidden * rsqrt-mean) -> gemm_bf16pre -> y.
// ---------------------------------------------------------------------------

#define DMODEL 512
#define BATCH  8
#define NTOK   4096
#define MROWS  (BATCH * NTOK)        // 32768
#define NELEM  ((size_t)MROWS * DMODEL)

#define TM 128          // workgroup M tile
#define TN 64           // workgroup N tile
#define KT 32           // K chunk (one bf16 WMMA K)
#define KPAD 40         // bf16 elems per LDS row: 80B pitch (64B data + 16B pad)

typedef __attribute__((ext_vector_type(16))) __bf16 bfx16;
typedef __attribute__((ext_vector_type(8)))  __bf16 bfx8;
typedef __attribute__((ext_vector_type(4)))  __bf16 bfx4;
typedef __attribute__((ext_vector_type(8)))  float  fx8;
typedef __attribute__((ext_vector_type(4)))  unsigned int u32x4;
typedef __attribute__((ext_vector_type(4)))  int i32x4;
typedef __attribute__((ext_vector_type(8)))  int i32x8;

#if defined(__has_builtin)
#if __has_builtin(__builtin_amdgcn_tensor_load_to_lds) && __has_builtin(__builtin_amdgcn_s_wait_tensorcnt)
#define HAVE_TDM 1
#endif
#endif
#ifndef HAVE_TDM
#define HAVE_TDM 0
#endif

// ---- bf16 helpers (explicit RNE) ------------------------------------------
__device__ __forceinline__ __bf16 f2bf(float f) {
    union { float f; unsigned u; } v; v.f = f;
    unsigned r = v.u + 0x7fffu + ((v.u >> 16) & 1u);   // round-to-nearest-even
    unsigned short h = (unsigned short)(r >> 16);
    return __builtin_bit_cast(__bf16, h);
}
__device__ __forceinline__ float bf2f(__bf16 b) {
    unsigned short h = __builtin_bit_cast(unsigned short, b);
    union { unsigned u; float f; } v; v.u = ((unsigned)h) << 16;
    return v.f;
}
struct bfpair { __bf16 h, l; };
__device__ __forceinline__ bfpair split2(float f) {
    bfpair p;
    p.h = f2bf(f);
    p.l = f2bf(f - bf2f(p.h));
    return p;
}

// A fragment (16-bit A 16x32): per-lane K groups [kq..kq+7] and [kq+16..kq+23]
__device__ __forceinline__ bfx16 frag_gap8(const __bf16* p) {
    bfx8 a = *(const bfx8*)(p);
    bfx8 b = *(const bfx8*)(p + 16);
    return __builtin_shufflevector(a, b, 0,1,2,3,4,5,6,7,8,9,10,11,12,13,14,15);
}
// contiguous-K fragment: per-lane K [kb..kb+15]; unused, B is loaded with frag_gap8
__device__ __forceinline__ bfx16 frag_contig16(const __bf16* p) {
    bfx8 a = *(const bfx8*)(p);
    bfx8 b = *(const bfx8*)(p + 8);
    return __builtin_shufflevector(a, b, 0,1,2,3,4,5,6,7,8,9,10,11,12,13,14,15);
}

#if HAVE_TDM
// ---------------------------------------------------------------------------
// Issue one TDM 2-D tile load: tile (tile_d1 rows x 32 bf16) from a row-major
// [tensor_d1 x 512] bf16 tensor into LDS at lds_off, with 16B padding after
// every 64B row (pad_interval = 16 dwords, pad_amount = 4 dwords) so the LDS
// pitch is KPAD(=40) bf16.  D# fields per CDNA5 ISA 08_async_tensor §8.3-8.5.
// This toolchain exposes the 6-arg builtin:
//   (u32x4 g0, i32x8 g1, i32x4 g2, i32x4 g3, i32x8 extra, i32 cpol)
// ---------------------------------------------------------------------------
__device__ __forceinline__ void tdm_load_tile(unsigned lds_off, const __bf16* gaddr,
                                              unsigned tensor_d1, unsigned tile_d1) {
    unsigned long long ga = (unsigned long long)(uintptr_t)gaddr;
    u32x4 g0;
    g0.x = 1u;                                                  // count=1, user mode
    g0.y = lds_off;                                             // lds_addr (bytes)
    g0.z = (unsigned)ga;                                        // global_addr[31:0]
    g0.w = (unsigned)((ga >> 32) & 0x01ffffffu) | (2u << 30);   // addr[56:32], type=2
    i32x8 g1;
    g1[0] = (int)((1u << 16)      // data_size: 2 bytes
                | (1u << 20)      // pad_enable
                | (3u << 22)      // pad_interval: 16 dwords (64B)
                | (3u << 25));    // pad_amount:   4 dwords (16B)
    g1[1] = (int)(512u << 16);                                  // tensor_dim0 = 512
    g1[2] = (int)((tensor_d1 & 0xffffu) << 16);                 // tensor_dim1 lo16
    g1[3] = (int)(((tensor_d1 >> 16) & 0xffffu) | (32u << 16)); // dim1 hi16, tile_dim0=32
    g1[4] = (int)(tile_d1 & 0xffffu);                           // tile_dim1, tile_dim2=0
    g1[5] = 512;                                                // tensor_dim0_stride
    g1[6] = 0;
    g1[7] = 0;
    i32x4 gz4 = {0, 0, 0, 0};
    i32x8 gz8 = {0, 0, 0, 0, 0, 0, 0, 0};
    __builtin_amdgcn_tensor_load_to_lds(g0, g1, gz4, gz4, gz8, 0);
}
#endif

// fallback / non-TDM tile copy: rows x 32 bf16, LDS pitch KPAD
__device__ __forceinline__ void copy_tile(const __bf16* __restrict__ g,
                                          __bf16* __restrict__ lds,
                                          int rows, int tid) {
    for (int i = tid; i < rows * 4; i += 256) {      // 16B chunks, 4 per row
        int r = i >> 2, c = (i & 3) * 8;
        *(bfx8*)(lds + r * KPAD + c) = *(const bfx8*)(g + (size_t)r * DMODEL + c);
    }
}

// ---------------------------------------------------------------------------
// Split-bf16 GEMM on pre-split operands:
//   out[M x 512] = act( (Ahi+Alo) @ (Bhi+Blo)^T' + bias )
// A*: [MROWS x 512] bf16 row-major.  B*: [512 x 512] bf16, row = output col n,
// contiguous K.  act: 0 = silu, 1 = sigmoid, 2 = identity.
// ---------------------------------------------------------------------------
__global__ __launch_bounds__(256)
void gemm_bf16pre(const __bf16* __restrict__ Ahi, const __bf16* __restrict__ Alo,
                  const __bf16* __restrict__ Bhi, const __bf16* __restrict__ Blo,
                  const float* __restrict__ bias, float* __restrict__ out, int act) {
    __shared__ __align__(16) __bf16 sA[2][2][TM][KPAD];   // [buf][hi/lo]
    __shared__ __align__(16) __bf16 sB[2][2][TN][KPAD];

    const int tid  = threadIdx.x;
    const int lane = tid & 31;
    const int wave = tid >> 5;
    const int wm   = wave >> 1;        // 0..3
    const int wn   = wave & 1;         // 0..1
    const int m0   = blockIdx.y * TM;
    const int n0   = blockIdx.x * TN;
    const int NKT  = DMODEL / KT;      // 16

    fx8 acc[2][2];
    #pragma unroll
    for (int mi = 0; mi < 2; ++mi)
        #pragma unroll
        for (int ni = 0; ni < 2; ++ni)
            #pragma unroll
            for (int e = 0; e < 8; ++e) acc[mi][ni][e] = 0.0f;

    const int rowF = lane & 15;
    const int kqA  = (lane >> 4) * 8;
    const int kqB  = (lane >> 4) * 8;

#if HAVE_TDM
    // ---- issue all 4 tile DMAs for one k-chunk (wave 0 only) ----
    auto issue_tile = [&](int buf, int k0) {
        tdm_load_tile((unsigned)(uintptr_t)&sA[buf][0][0][0],
                      Ahi + (size_t)m0 * DMODEL + k0, MROWS, TM);
        tdm_load_tile((unsigned)(uintptr_t)&sA[buf][1][0][0],
                      Alo + (size_t)m0 * DMODEL + k0, MROWS, TM);
        tdm_load_tile((unsigned)(uintptr_t)&sB[buf][0][0][0],
                      Bhi + (size_t)n0 * DMODEL + k0, DMODEL, TN);
        tdm_load_tile((unsigned)(uintptr_t)&sB[buf][1][0][0],
                      Blo + (size_t)n0 * DMODEL + k0, DMODEL, TN);
    };
    if (wave == 0) {
        issue_tile(0, 0);
        __builtin_amdgcn_s_wait_tensorcnt(0);
    }
    __syncthreads();
#endif

    for (int t = 0; t < NKT; ++t) {
        const int cur = t & 1;
#if HAVE_TDM
        // prefetch tile t+1 into the other buffer while computing on cur
        if (wave == 0 && t + 1 < NKT) issue_tile(cur ^ 1, (t + 1) * KT);
#else
        copy_tile(Ahi + (size_t)m0 * DMODEL + t * KT, &sA[cur][0][0][0], TM, tid);
        copy_tile(Alo + (size_t)m0 * DMODEL + t * KT, &sA[cur][1][0][0], TM, tid);
        copy_tile(Bhi + (size_t)n0 * DMODEL + t * KT, &sB[cur][0][0][0], TN, tid);
        copy_tile(Blo + (size_t)n0 * DMODEL + t * KT, &sB[cur][1][0][0], TN, tid);
        __syncthreads();
#endif
        bfx16 aHi[2], aLo[2], bHi[2], bLo[2];
        #pragma unroll
        for (int mi = 0; mi < 2; ++mi) {
            const __bf16* ph = &sA[cur][0][wm * 32 + mi * 16 + rowF][0];
            const __bf16* pl = &sA[cur][1][wm * 32 + mi * 16 + rowF][0];
            aHi[mi] = frag_gap8(ph + kqA);
            aLo[mi] = frag_gap8(pl + kqA);
        }
        #pragma unroll
        for (int ni = 0; ni < 2; ++ni) {
            const __bf16* ph = &sB[cur][0][wn * 32 + ni * 16 + rowF][0];
            const __bf16* pl = &sB[cur][1][wn * 32 + ni * 16 + rowF][0];
            bHi[ni] = frag_gap8(ph + kqB);
            bLo[ni] = frag_gap8(pl + kqB);
        }
        #pragma unroll
        for (int mi = 0; mi < 2; ++mi)
            #pragma unroll
            for (int ni = 0; ni < 2; ++ni) {
                acc[mi][ni] = __builtin_amdgcn_wmma_f32_16x16x32_bf16(
                    false, aHi[mi], false, bHi[ni], (short)0, acc[mi][ni], false, false);
                acc[mi][ni] = __builtin_amdgcn_wmma_f32_16x16x32_bf16(
                    false, aHi[mi], false, bLo[ni], (short)0, acc[mi][ni], false, false);
                acc[mi][ni] = __builtin_amdgcn_wmma_f32_16x16x32_bf16(
                    false, aLo[mi], false, bHi[ni], (short)0, acc[mi][ni], false, false);
            }
        __syncthreads();                 // everyone done reading buf[cur]
#if HAVE_TDM
        if (wave == 0 && t + 1 < NKT) __builtin_amdgcn_s_wait_tensorcnt(0);
        __syncthreads();                 // buf[cur^1] is ready for next iter
#endif
    }

    // ---- epilogue: C/D layout -> lane holds col n=lane%16, rows (lane/16)*8+e
    #pragma unroll
    for (int mi = 0; mi < 2; ++mi) {
        #pragma unroll
        for (int ni = 0; ni < 2; ++ni) {
            int n  = n0 + wn * 32 + ni * 16 + (lane & 15);
            int mb = m0 + wm * 32 + mi * 16 + (lane >> 4) * 8;
            float bv = bias[n];
            #pragma unroll
            for (int e = 0; e < 8; ++e) {
                float v = acc[mi][ni][e] + bv;
                if (act == 0)      v = v / (1.0f + expf(-v));    // silu
                else if (act == 1) v = 1.0f / (1.0f + expf(-v)); // sigmoid
                out[(size_t)(mb + e) * DMODEL + n] = v;
            }
        }
    }
}

// ---------------------------------------------------------------------------
// fp32 -> (hi,lo) bf16 split, optionally fused with output-gate * rmsnorm:
//   src' = src * gate * rnorm[row]   (when gate != nullptr)
// ---------------------------------------------------------------------------
__global__ __launch_bounds__(256)
void split_kernel(const float* __restrict__ src,
                  const float* __restrict__ gate,
                  const float* __restrict__ rnorm,
                  __bf16* __restrict__ hi, __bf16* __restrict__ lo) {
    const size_t i = ((size_t)blockIdx.x * 256 + threadIdx.x) * 4;
    float4 v = *(const float4*)(src + i);
    if (gate != nullptr) {
        float4 g = *(const float4*)(gate + i);
        float rn = rnorm[i >> 9];                 // row = i / DMODEL
        v.x *= g.x * rn; v.y *= g.y * rn; v.z *= g.z * rn; v.w *= g.w * rn;
    }
    bfx4 h4, l4;
    bfpair p0 = split2(v.x); h4[0] = p0.h; l4[0] = p0.l;
    bfpair p1 = split2(v.y); h4[1] = p1.h; l4[1] = p1.l;
    bfpair p2 = split2(v.z); h4[2] = p2.h; l4[2] = p2.l;
    bfpair p3 = split2(v.w); h4[3] = p3.h; l4[3] = p3.l;
    *(bfx4*)(hi + i) = h4;
    *(bfx4*)(lo + i) = l4;
}

// weight split + transpose: W[k][n] fp32 -> Wt*[n][k] bf16 (hi, lo)
__global__ __launch_bounds__(256)
void splitw_kernel(const float* __restrict__ W,
                   __bf16* __restrict__ hi, __bf16* __restrict__ lo) {
    const int i = blockIdx.x * 256 + threadIdx.x;   // 0 .. 512*512-1
    const int k = i >> 9, n = i & 511;
    bfpair p = split2(W[i]);
    hi[(size_t)n * DMODEL + k] = p.h;
    lo[(size_t)n * DMODEL + k] = p.l;
}

// ---------------------------------------------------------------------------
// Bidirectional complex gated scan along one grid axis (length 64).
// One block per (b, fixed-other-axis) line; thread = channel d.
// ---------------------------------------------------------------------------
__global__ __launch_bounds__(DMODEL)
void scan2d_kernel(const float* __restrict__ U, const float* __restrict__ F,
                   const float* __restrict__ lb_ptr,
                   const float* __restrict__ theta,
                   float* __restrict__ HID, int axis, int init) {
    const int line = blockIdx.x;        // 0 .. B*64-1
    const int b    = line >> 6;
    const int pos  = line & 63;
    const int d    = threadIdx.x;
    const float lb = lb_ptr[0];
    const float th = theta[d];
    const float c  = cosf(th);
    const float s  = sinf(th);

    size_t base, stride;
    if (axis == 1) {  // scan along w: n = pos*64 + t
        base   = ((size_t)(b * NTOK + pos * 64)) * DMODEL + d;
        stride = DMODEL;
    } else {          // scan along h: n = t*64 + pos
        base   = ((size_t)(b * NTOK + pos)) * DMODEL + d;
        stride = (size_t)64 * DMODEL;
    }

    float hr = 0.0f, hi = 0.0f;
    for (int t = 0; t < 64; ++t) {
        size_t idx = base + (size_t)t * stride;
        float f   = F[idx];
        float u   = U[idx];
        float lam = lb + (1.0f - lb) * f;
        float dr  = lam * c, di = lam * s;
        float uin = (1.0f - lam) * u;
        float nr  = fmaf(dr, hr, fmaf(-di, hi, uin));
        float ni  = fmaf(dr, hi, di * hr);
        hr = nr; hi = ni;
        if (init) HID[idx] = hr;
        else      HID[idx] += hr;
    }
    hr = 0.0f; hi = 0.0f;
    for (int t = 63; t >= 0; --t) {
        size_t idx = base + (size_t)t * stride;
        float f   = F[idx];
        float u   = U[idx];
        float lam = lb + (1.0f - lb) * f;
        float dr  = lam * c, di = lam * s;
        float uin = (1.0f - lam) * u;
        float nr  = fmaf(dr, hr, fmaf(-di, hi, uin));
        float ni  = fmaf(dr, hi, di * hr);
        hr = nr; hi = ni;
        HID[idx] += hr;
    }
}

// ---------------------------------------------------------------------------
// Per-row rsqrt(mean(h^2)+eps): one wave32 per row, 8 rows per block.
// ---------------------------------------------------------------------------
__global__ __launch_bounds__(256)
void rownorm_kernel(const float* __restrict__ H, float* __restrict__ RN) {
    const int row  = blockIdx.x * 8 + (threadIdx.x >> 5);
    const int lane = threadIdx.x & 31;
    const float* p = H + (size_t)row * DMODEL;
    float ssum = 0.0f;
    #pragma unroll
    for (int i = lane; i < DMODEL; i += 32) {
        float v = p[i];
        ssum = fmaf(v, v, ssum);
    }
    #pragma unroll
    for (int off = 16; off > 0; off >>= 1)
        ssum += __shfl_xor(ssum, off, 32);
    if (lane == 0)
        RN[row] = rsqrtf(ssum * (1.0f / (float)DMODEL) + 1e-6f);
}

// ---------------------------------------------------------------------------
extern "C" void kernel_launch(void* const* d_in, const int* in_sizes, int n_in,
                              void* d_out, int out_size, void* d_ws, size_t ws_size,
                              hipStream_t stream) {
    const float* x     = (const float*)d_in[0];
    const float* lb    = (const float*)d_in[1];
    const float* W_in  = (const float*)d_in[2];
    const float* b_in  = (const float*)d_in[3];
    const float* W_f   = (const float*)d_in[4];
    const float* b_f   = (const float*)d_in[5];
    const float* theta = (const float*)d_in[6];
    const float* W_g   = (const float*)d_in[7];
    const float* b_g   = (const float*)d_in[8];
    const float* W_out = (const float*)d_in[9];
    const float* b_out = (const float*)d_in[10];
    float* out = (float*)d_out;

    // workspace layout
    float*  U    = (float*)d_ws;
    float*  F    = U + NELEM;
    float*  G    = F + NELEM;
    float*  HID  = G + NELEM;
    float*  RN   = HID + NELEM;                 // MROWS floats
    __bf16* Xhi  = (__bf16*)(RN + MROWS);       // reused for Z in phase 3
    __bf16* Xlo  = Xhi + NELEM;
    __bf16* Wbuf = Xlo + NELEM;                 // 8 x [512*512] bf16
    const size_t WSZ = (size_t)DMODEL * DMODEL;
    __bf16* Wih = Wbuf;            __bf16* Wil = Wbuf + WSZ;
    __bf16* Wfh = Wbuf + 2 * WSZ;  __bf16* Wfl = Wbuf + 3 * WSZ;
    __bf16* Wgh = Wbuf + 4 * WSZ;  __bf16* Wgl = Wbuf + 5 * WSZ;
    __bf16* Woh = Wbuf + 6 * WSZ;  __bf16* Wol = Wbuf + 7 * WSZ;

    dim3 gblock(256);
    dim3 ggrid(DMODEL / TN, MROWS / TM, 1);     // (8, 256)
    const int wgrid = (DMODEL * DMODEL) / 256;  // 1024
    const int sgrid = (int)(NELEM / 4 / 256);   // 16384

    // Phase 0: one-time hi/lo splits (weights transposed to [n][k])
    splitw_kernel<<<wgrid, gblock, 0, stream>>>(W_in,  Wih, Wil);
    splitw_kernel<<<wgrid, gblock, 0, stream>>>(W_f,   Wfh, Wfl);
    splitw_kernel<<<wgrid, gblock, 0, stream>>>(W_g,   Wgh, Wgl);
    splitw_kernel<<<wgrid, gblock, 0, stream>>>(W_out, Woh, Wol);
    split_kernel<<<sgrid, gblock, 0, stream>>>(x, nullptr, nullptr, Xhi, Xlo);

    // Phase 1: three input GEMMs with fused activations
    gemm_bf16pre<<<ggrid, gblock, 0, stream>>>(Xhi, Xlo, Wih, Wil, b_in, U, 0);
    gemm_bf16pre<<<ggrid, gblock, 0, stream>>>(Xhi, Xlo, Wfh, Wfl, b_f,  F, 1);
    gemm_bf16pre<<<ggrid, gblock, 0, stream>>>(Xhi, Xlo, Wgh, Wgl, b_g,  G, 1);

    // Phase 2: four directional scans over the 64x64 grid
    scan2d_kernel<<<dim3(BATCH * 64), dim3(DMODEL), 0, stream>>>(U, F, lb, theta, HID, 1, 1);
    scan2d_kernel<<<dim3(BATCH * 64), dim3(DMODEL), 0, stream>>>(U, F, lb, theta, HID, 0, 0);

    // Phase 3: row RMS factors, gated-normalized split (reusing X buffers),
    //          then the output GEMM
    rownorm_kernel<<<dim3(MROWS / 8), dim3(256), 0, stream>>>(HID, RN);
    split_kernel<<<sgrid, gblock, 0, stream>>>(HID, G, RN, Xhi, Xlo);
    gemm_bf16pre<<<ggrid, gblock, 0, stream>>>(Xhi, Xlo, Woh, Wol, b_out, out, 2);
}
